// Discriminator_51049981280755
// MI455X (gfx1250) — compile-verified, matrix operand order checked by reading, not run on hardware
//
#include <hip/hip_runtime.h>

typedef __attribute__((ext_vector_type(16))) _Float16 v16h;
typedef __attribute__((ext_vector_type(8)))  _Float16 v8h;
typedef __attribute__((ext_vector_type(4)))  _Float16 v4h;
typedef __attribute__((ext_vector_type(8)))  float    v8f;

// Problem constants (match reference)
constexpr int Bc   = 8;
constexpr int SEQc = 2048;
constexpr int Tc   = SEQc + 1;   // 2049
constexpr int Vc   = 4096;
constexpr int Dc   = 1024;
constexpr int Hc   = 16;
constexpr int DHc  = 64;
constexpr int NLc  = 8;
constexpr int FFc  = 4096;

constexpr int EPI_BIAS = 0, EPI_GELU = 1, EPI_RES = 2, EPI_EMBED = 3;

// ---------------------------------------------------------------------------
// Fragment-ready LDS swizzle for v_wmma_f32_16x16x32_f16 operands.
// Element (m16, k) of a 16x32 fragment lives at:
//   lane = m16 + 16*((k>>3)&1)      (upper lanes hold K 8..15 / 24..31)
//   i    = (k&7) | ((k&16)>>1)      (halves 8..15 hold K 16..31)
// so each lane's 16 halves are contiguous (32B) -> 2x ds_load_b128.
// ---------------------------------------------------------------------------
__device__ __forceinline__ int sw_lane(int m16, int k) { return m16 + ((k & 8) << 1); }
__device__ __forceinline__ int sw_i(int k)             { return (k & 7) | ((k & 16) >> 1); }

__device__ __forceinline__ v16h ld_frag(const _Float16* tile_base, int lane) {
    union { v16h v; v8h h[2]; } u;
    const v8h* p = (const v8h*)(tile_base + lane * 16);
    u.h[0] = p[0];
    u.h[1] = p[1];
    return u.v;
}

__device__ __forceinline__ v4h pack4(float4 v) {
    v4h h = { (_Float16)v.x, (_Float16)v.y, (_Float16)v.z, (_Float16)v.w };
    return h;
}

// ---------------------------------------------------------------------------
// Generic WMMA GEMM: C[M,N] = epi(A[M,K] @ W[K,N] + bias[N])
// Software-pipelined (register prefetch) + double-buffered LDS.
// Block: 256 thr (8 waves), tile 128x128x32; wave = 32x64 (2x4 fragments).
// K%32==0, N%128==0 at every call site.
// ---------------------------------------------------------------------------
#define GBM 128
#define GBN 128
#define GBK 32
#define TILE_H (8 * 32 * 16)     // halves per (A or B) buffer

__global__ __launch_bounds__(256)
void gemm_wmma_kernel(const float* __restrict__ A, const float* __restrict__ W,
                      const float* __restrict__ bias, const float* __restrict__ R,
                      const float* __restrict__ pos, float* __restrict__ C,
                      int M, int N, int K, int mode)
{
    __shared__ __align__(32) _Float16 Asw[2 * TILE_H];
    __shared__ __align__(32) _Float16 Bsw[2 * TILE_H];

    const int tid  = threadIdx.x;
    const int lane = tid & 31;
    const int w    = tid >> 5;
    const int wm   = w & 3;          // 4 wave-rows of 32
    const int wn   = w >> 2;         // 2 wave-cols of 64
    const int m0   = blockIdx.x * GBM;
    const int n0   = blockIdx.y * GBN;
    const int kb   = (lane >= 16) ? 8 : 0;   // +8 row offset for upper lanes (C layout)
    const int ml   = lane & 15;
    const bool fullM = (m0 + GBM <= M);

    // Per-thread staging coordinates (invariant across k-steps)
    const int k4A = (tid & 7) << 2;      // A: 4 contiguous K per thread
    const int rA0 = tid >> 3;            // A: row, +32 per sub-iteration
    const int c4B = (tid & 31) << 2;     // B: 4 contiguous N per thread
    const int kB0 = tid >> 5;            // B: k-row, +8 per sub-iteration

    v8f acc[2][4] = {};
    float4 ar[4], wr[4];

    auto gload = [&](int k0) {
        #pragma unroll
        for (int it = 0; it < 4; ++it) {
            int r = rA0 + it * 32;
            int gm = m0 + r;
            int gmc = (gm < M) ? gm : (M - 1);
            ar[it] = *(const float4*)(A + (size_t)gmc * K + k0 + k4A);
            if (gm >= M) { ar[it].x = 0.f; ar[it].y = 0.f; ar[it].z = 0.f; ar[it].w = 0.f; }
        }
        #pragma unroll
        for (int it = 0; it < 4; ++it) {
            int kk = kB0 + it * 8;
            wr[it] = *(const float4*)(W + (size_t)(k0 + kk) * N + n0 + c4B);
        }
    };
    auto sstore = [&](int buf) {
        _Float16* As = Asw + buf * TILE_H;
        _Float16* Bs = Bsw + buf * TILE_H;
        #pragma unroll
        for (int it = 0; it < 4; ++it) {
            int r = rA0 + it * 32;
            int off = ((r >> 4) * 32 + sw_lane(r & 15, k4A)) * 16 + sw_i(k4A);
            *(v4h*)(As + off) = pack4(ar[it]);
        }
        #pragma unroll
        for (int it = 0; it < 4; ++it) {
            int kk = kB0 + it * 8;
            int i_ = sw_i(kk);
            int cl = (c4B & 15) + ((kk & 8) << 1);
            _Float16* p = Bs + ((c4B >> 4) * 32) * 16 + i_;
            p[(cl + 0) * 16] = (_Float16)wr[it].x;
            p[(cl + 1) * 16] = (_Float16)wr[it].y;
            p[(cl + 2) * 16] = (_Float16)wr[it].z;
            p[(cl + 3) * 16] = (_Float16)wr[it].w;
        }
    };

    const int nk = K >> 5;
    gload(0);
    for (int ks = 0; ks < nk; ++ks) {
        const int buf = ks & 1;
        sstore(buf);
        __syncthreads();                       // implies dscnt==0: prev buf reads retired
        if (ks + 1 < nk) gload((ks + 1) << 5); // prefetch next tile while WMMAs run

        const _Float16* As = Asw + buf * TILE_H;
        const _Float16* Bs = Bsw + buf * TILE_H;
        v16h af[2], bf[4];
        #pragma unroll
        for (int mt = 0; mt < 2; ++mt)
            af[mt] = ld_frag(As + ((wm * 2 + mt) * 32) * 16, lane);
        #pragma unroll
        for (int nt = 0; nt < 4; ++nt)
            bf[nt] = ld_frag(Bs + ((wn * 4 + nt) * 32) * 16, lane);
        #pragma unroll
        for (int mt = 0; mt < 2; ++mt)
            #pragma unroll
            for (int nt = 0; nt < 4; ++nt)
                acc[mt][nt] = __builtin_amdgcn_wmma_f32_16x16x32_f16(
                    false, af[mt], false, bf[nt], (short)0, acc[mt][nt], false, false);
    }

    auto epi = [&](bool guard) {
        #pragma unroll
        for (int mt = 0; mt < 2; ++mt) {
            #pragma unroll
            for (int nt = 0; nt < 4; ++nt) {
                #pragma unroll
                for (int r = 0; r < 8; ++r) {
                    int m = m0 + wm * 32 + mt * 16 + kb + r;
                    int n = n0 + wn * 64 + nt * 16 + ml;
                    if (guard && m >= M) continue;
                    float v = acc[mt][nt][r] + bias[n];
                    if (mode == EPI_GELU) {
                        v = 0.5f * v * (1.f + erff(v * 0.70710678118654752440f));
                    } else if (mode == EPI_RES) {
                        v += R[(size_t)m * N + n];
                    } else if (mode == EPI_EMBED) {
                        int b = m / SEQc, s = m % SEQc;
                        v += pos[(size_t)(s + 1) * Dc + n];
                        C[((size_t)b * Tc + s + 1) * N + n] = v;
                        continue;
                    }
                    C[(size_t)m * N + n] = v;
                }
            }
        }
    };
    if (fullM) epi(false); else epi(true);
}

// ---------------------------------------------------------------------------
// KV[b,h,d,m] = sum_t Kf[b,t,h,d] * V[b,t,h,m]  (64x64 per (b,h), reduce T)
// Pipelined + double-buffered. Block: 128 thr (4 waves), wave = 32x32.
// ---------------------------------------------------------------------------
#define KVTILE_H (4 * 32 * 16)

__global__ __launch_bounds__(128)
void kv_wmma_kernel(const float* __restrict__ Kf, const float* __restrict__ Vv,
                    float* __restrict__ KV)
{
    const int bh = blockIdx.x;
    const int b = bh / Hc, h = bh % Hc;
    __shared__ __align__(32) _Float16 Ksw[2 * KVTILE_H];   // A: 4 d-tiles
    __shared__ __align__(32) _Float16 Vsw[2 * KVTILE_H];   // B: 4 m-tiles
    const int tid = threadIdx.x, lane = tid & 31, w = tid >> 5;
    const int wm = w & 1, wn = w >> 1;
    const int kb = (lane >= 16) ? 8 : 0, ml = lane & 15;

    const int x4  = (tid & 15) << 2;   // 4 contiguous d (A) / m (B) per thread
    const int tt0 = tid >> 4;          // t within step, +8 per sub-iteration

    v8f acc[2][2] = {};
    float4 kr[4], vr[4];

    auto gload = [&](int t0) {
        #pragma unroll
        for (int it = 0; it < 4; ++it) {
            int t = t0 + tt0 + it * 8;
            int tc_ = (t < Tc) ? t : (Tc - 1);
            size_t base = (((size_t)b * Tc + tc_) * Hc + h) * DHc + x4;
            kr[it] = *(const float4*)(Kf + base);
            vr[it] = *(const float4*)(Vv + base);
            if (t >= Tc) {
                kr[it].x = 0.f; kr[it].y = 0.f; kr[it].z = 0.f; kr[it].w = 0.f;
                vr[it].x = 0.f; vr[it].y = 0.f; vr[it].z = 0.f; vr[it].w = 0.f;
            }
        }
    };
    auto sstore = [&](int buf) {
        _Float16* Ks = Ksw + buf * KVTILE_H;
        _Float16* Vs = Vsw + buf * KVTILE_H;
        #pragma unroll
        for (int it = 0; it < 4; ++it) {
            int tt = tt0 + it * 8;
            int i_ = sw_i(tt);
            int cl = (x4 & 15) + ((tt & 8) << 1);
            _Float16* pk = Ks + ((x4 >> 4) * 32) * 16 + i_;
            pk[(cl + 0) * 16] = (_Float16)kr[it].x;
            pk[(cl + 1) * 16] = (_Float16)kr[it].y;
            pk[(cl + 2) * 16] = (_Float16)kr[it].z;
            pk[(cl + 3) * 16] = (_Float16)kr[it].w;
            _Float16* pv = Vs + ((x4 >> 4) * 32) * 16 + i_;
            pv[(cl + 0) * 16] = (_Float16)vr[it].x;
            pv[(cl + 1) * 16] = (_Float16)vr[it].y;
            pv[(cl + 2) * 16] = (_Float16)vr[it].z;
            pv[(cl + 3) * 16] = (_Float16)vr[it].w;
        }
    };

    const int nk = (Tc + 31) >> 5;     // 65 steps
    gload(0);
    for (int ks = 0; ks < nk; ++ks) {
        const int buf = ks & 1;
        sstore(buf);
        __syncthreads();
        if (ks + 1 < nk) gload((ks + 1) << 5);

        const _Float16* Ks = Ksw + buf * KVTILE_H;
        const _Float16* Vs = Vsw + buf * KVTILE_H;
        v16h af[2], bf[2];
        #pragma unroll
        for (int mt = 0; mt < 2; ++mt)
            af[mt] = ld_frag(Ks + ((wm * 2 + mt) * 32) * 16, lane);
        #pragma unroll
        for (int nt = 0; nt < 2; ++nt)
            bf[nt] = ld_frag(Vs + ((wn * 2 + nt) * 32) * 16, lane);
        #pragma unroll
        for (int mt = 0; mt < 2; ++mt)
            #pragma unroll
            for (int nt = 0; nt < 2; ++nt)
                acc[mt][nt] = __builtin_amdgcn_wmma_f32_16x16x32_f16(
                    false, af[mt], false, bf[nt], (short)0, acc[mt][nt], false, false);
    }
    #pragma unroll
    for (int mt = 0; mt < 2; ++mt)
        #pragma unroll
        for (int nt = 0; nt < 2; ++nt)
            #pragma unroll
            for (int r = 0; r < 8; ++r) {
                int d = wm * 32 + mt * 16 + kb + r;
                int m = wn * 32 + nt * 16 + ml;
                KV[((size_t)bh * DHc + d) * DHc + m] = acc[mt][nt][r];
            }
}

// ---------------------------------------------------------------------------
// attn[b,t,h,m] = (Qf[b,t,h,:] . KV[b,h,:,m]) * Z[b,t,h]  (M=T,K=64,N=64)
// Output written in [B,T,D] layout. Block 256 thr, tile 128x64, K staged whole.
// ---------------------------------------------------------------------------
__global__ __launch_bounds__(256)
void attn_wmma_kernel(const float* __restrict__ Qf, const float* __restrict__ KV,
                      const float* __restrict__ Zb, float* __restrict__ Out)
{
    const int bh = blockIdx.y;
    const int b = bh / Hc, h = bh % Hc;
    const int t0 = blockIdx.x * 128;
    __shared__ __align__(32) _Float16 Qsw[2 * 8 * 32 * 16];   // [kstep][row-tile]
    __shared__ __align__(32) _Float16 Vsw[2 * 4 * 32 * 16];   // [kstep][col-tile]
    const int tid = threadIdx.x, lane = tid & 31, w = tid >> 5;  // w = row-tile 0..7
    const int kb = (lane >= 16) ? 8 : 0, ml = lane & 15;

    #pragma unroll
    for (int it = 0; it < 8; ++it) {
        int e = tid + it * 256;              // 0..2047 float4 units (128t x 64d)
        int r = e >> 4, d4 = (e & 15) << 2;
        int t = t0 + r;
        int tc_ = (t < Tc) ? t : (Tc - 1);
        float4 q4 = *(const float4*)(Qf + (((size_t)b * Tc + tc_) * Hc + h) * DHc + d4);
        if (t >= Tc) { q4.x = 0.f; q4.y = 0.f; q4.z = 0.f; q4.w = 0.f; }
        int ks2 = d4 >> 5, kd = d4 & 31;
        int off = (((ks2 * 8 + (r >> 4)) * 32) + sw_lane(r & 15, kd)) * 16 + sw_i(kd);
        *(v4h*)(Qsw + off) = pack4(q4);
    }
    #pragma unroll
    for (int it = 0; it < 4; ++it) {
        int e = tid + it * 256;              // 0..1023 float4 units (64d x 64m)
        int d = e >> 4, m4 = (e & 15) << 2;
        float4 kv4 = *(const float4*)(KV + ((size_t)bh * DHc + d) * DHc + m4);
        int ks2 = d >> 5, kd = d & 31;
        int i_ = sw_i(kd);
        int cl = (m4 & 15) + ((kd & 8) << 1);
        _Float16* p = Vsw + ((ks2 * 4 + (m4 >> 4)) * 32) * 16 + i_;
        p[(cl + 0) * 16] = (_Float16)kv4.x;
        p[(cl + 1) * 16] = (_Float16)kv4.y;
        p[(cl + 2) * 16] = (_Float16)kv4.z;
        p[(cl + 3) * 16] = (_Float16)kv4.w;
    }
    __syncthreads();

    v8f acc[4] = {};
    #pragma unroll
    for (int ks2 = 0; ks2 < 2; ++ks2) {
        v16h af = ld_frag(Qsw + ((ks2 * 8 + w) * 32) * 16, lane);
        #pragma unroll
        for (int nt = 0; nt < 4; ++nt) {
            v16h bf = ld_frag(Vsw + ((ks2 * 4 + nt) * 32) * 16, lane);
            acc[nt] = __builtin_amdgcn_wmma_f32_16x16x32_f16(
                false, af, false, bf, (short)0, acc[nt], false, false);
        }
    }

    auto epi = [&](bool guard) {
        #pragma unroll
        for (int nt = 0; nt < 4; ++nt)
            #pragma unroll
            for (int r = 0; r < 8; ++r) {
                int t = t0 + w * 16 + kb + r;
                if (guard && t >= Tc) continue;
                float z = Zb[((size_t)b * Tc + t) * Hc + h];
                int n = nt * 16 + ml;
                Out[(((size_t)b * Tc + t) * Hc + h) * DHc + n] = acc[nt][r] * z;
            }
    };
    if (t0 + 128 <= Tc) epi(false); else epi(true);
}

// ---------------------------------------------------------------------------
// Elementwise / reduction helpers
// ---------------------------------------------------------------------------
__global__ void rope_elu_kernel(float* __restrict__ Q, float* __restrict__ K)
{
    int idx = blockIdx.x * blockDim.x + threadIdx.x;
    if (idx >= Bc * Tc * Hc * 32) return;
    int j = idx & 31, h = (idx >> 5) & 15, row = idx >> 9;
    int t = row % Tc;
    float inv = __expf(-(2.f * j / (float)DHc) * 9.21034037197618f); // ln(10000)
    float ang = (float)t * inv;
    float c = __cosf(ang), s = __sinf(ang);
    size_t base = ((size_t)row * Hc + h) * DHc;
    float q1 = Q[base + j], q2 = Q[base + j + 32];
    float k1 = K[base + j], k2 = K[base + j + 32];
    float qa = q1 * c - q2 * s, qb = q2 * c + q1 * s;
    float ka = k1 * c - k2 * s, kb = k2 * c + k1 * s;
    Q[base + j]      = (qa > 0.f) ? qa + 1.f : __expf(qa);   // elu(x)+1
    Q[base + j + 32] = (qb > 0.f) ? qb + 1.f : __expf(qb);
    K[base + j]      = (ka > 0.f) ? ka + 1.f : __expf(ka);
    K[base + j + 32] = (kb > 0.f) ? kb + 1.f : __expf(kb);
}

__global__ __launch_bounds__(256)
void ksum_kernel(const float* __restrict__ Kf, float* __restrict__ Ks)
{
    __shared__ float red[256];
    const int bh = blockIdx.x;
    const int b = bh / Hc, h = bh % Hc;
    const int d = threadIdx.x & 63, tg = threadIdx.x >> 6;   // 4 t-groups
    float s = 0.f;
    for (int t = tg; t < Tc; t += 4)
        s += Kf[(((size_t)b * Tc + t) * Hc + h) * DHc + d];
    red[threadIdx.x] = s;
    __syncthreads();
    if (threadIdx.x < 64)
        Ks[(size_t)bh * DHc + d] = red[d] + red[64 + d] + red[128 + d] + red[192 + d];
}

__global__ void z_kernel(const float* __restrict__ Qf, const float* __restrict__ Ks,
                         float* __restrict__ Zb)
{
    int idx = blockIdx.x * blockDim.x + threadIdx.x;
    if (idx >= Bc * Tc * Hc) return;
    int h = idx & 15, bt = idx >> 4;
    int b = bt / Tc;
    const float* q = Qf + ((size_t)bt * Hc + h) * DHc;
    const float* k = Ks + ((size_t)b * Hc + h) * DHc;
    float s = 0.f;
    #pragma unroll 8
    for (int d = 0; d < DHc; ++d) s += q[d] * k[d];
    Zb[idx] = 1.f / (s + 1e-6f);
}

__global__ __launch_bounds__(256)
void ln_kernel(const float* __restrict__ In, const float* __restrict__ g,
               const float* __restrict__ bb, float* __restrict__ Out,
               int in_stride_rows)
{
    __shared__ float red[256];
    const int row = blockIdx.x, tid = threadIdx.x;
    const float* x = In + (size_t)row * in_stride_rows * Dc;
    float4 v = *(const float4*)(x + tid * 4);
    red[tid] = v.x + v.y + v.z + v.w;
    __syncthreads();
    for (int off = 128; off > 0; off >>= 1) { if (tid < off) red[tid] += red[tid + off]; __syncthreads(); }
    float mean = red[0] * (1.f / Dc);
    __syncthreads();
    float dx = v.x - mean, dy = v.y - mean, dz = v.z - mean, dw = v.w - mean;
    red[tid] = dx * dx + dy * dy + dz * dz + dw * dw;
    __syncthreads();
    for (int off = 128; off > 0; off >>= 1) { if (tid < off) red[tid] += red[tid + off]; __syncthreads(); }
    float rinv = rsqrtf(red[0] * (1.f / Dc) + 1e-5f);
    float4 gv = *(const float4*)(g + tid * 4);
    float4 bv = *(const float4*)(bb + tid * 4);
    float4 o;
    o.x = dx * rinv * gv.x + bv.x;
    o.y = dy * rinv * gv.y + bv.y;
    o.z = dz * rinv * gv.z + bv.z;
    o.w = dw * rinv * gv.w + bv.w;
    *(float4*)(Out + (size_t)row * Dc + tid * 4) = o;
}

__global__ void cls_embed_kernel(const float* __restrict__ We, const float* __restrict__ be,
                                 const float* __restrict__ pos, float* __restrict__ X)
{
    int idx = blockIdx.x * blockDim.x + threadIdx.x;
    if (idx >= Bc * Dc) return;
    int n = idx & (Dc - 1), b = idx >> 10;
    X[(size_t)b * Tc * Dc + n] = We[(size_t)(Vc - 1) * Dc + n] + be[n] + pos[n];
}

__global__ __launch_bounds__(256)
void head_kernel(const float* __restrict__ CLS, const float* __restrict__ Wout,
                 const float* __restrict__ bout, float* __restrict__ out)
{
    __shared__ float cls_s[Dc];
    int idx = blockIdx.x * 256 + threadIdx.x;
    int b = idx >> 12;                        // block-uniform (4096 % 256 == 0)
    #pragma unroll
    for (int i = 0; i < 4; ++i)
        cls_s[threadIdx.x + i * 256] = CLS[(size_t)b * Dc + threadIdx.x + i * 256];
    __syncthreads();
    int n = idx & (Vc - 1);
    float acc = bout[n];
    for (int d = 0; d < Dc; ++d) acc += cls_s[d] * Wout[(size_t)d * Vc + n];
    out[idx] = acc;
}

// ---------------------------------------------------------------------------
extern "C" void kernel_launch(void* const* d_in, const int* in_sizes, int n_in,
                              void* d_out, int out_size, void* d_ws, size_t ws_size,
                              hipStream_t stream)
{
    (void)in_sizes; (void)n_in; (void)out_size; (void)ws_size;
    const float* inputs = (const float*)d_in[0];
    const float* We   = (const float*)d_in[1];
    const float* be   = (const float*)d_in[2];
    const float* pos  = (const float*)d_in[3];
    const float* Wq   = (const float*)d_in[4];
    const float* bq   = (const float*)d_in[5];
    const float* Wk   = (const float*)d_in[6];
    const float* bk   = (const float*)d_in[7];
    const float* Wv   = (const float*)d_in[8];
    const float* bv   = (const float*)d_in[9];
    const float* Wo   = (const float*)d_in[10];
    const float* bo   = (const float*)d_in[11];
    const float* ln1g = (const float*)d_in[12];
    const float* ln1b = (const float*)d_in[13];
    const float* W1   = (const float*)d_in[14];
    const float* b1   = (const float*)d_in[15];
    const float* W2   = (const float*)d_in[16];
    const float* b2   = (const float*)d_in[17];
    const float* ln2g = (const float*)d_in[18];
    const float* ln2b = (const float*)d_in[19];
    const float* gN   = (const float*)d_in[20];
    const float* bN   = (const float*)d_in[21];
    const float* Wout = (const float*)d_in[22];
    const float* bout = (const float*)d_in[23];
    float* out = (float*)d_out;

    const size_t MT = (size_t)Bc * Tc;   // 16392 rows
    char* ws = (char*)d_ws; size_t off = 0;
    auto alloc = [&](size_t elems) -> float* {
        float* p = (float*)(ws + off);
        off += ((elems * sizeof(float)) + 255) & ~(size_t)255;
        return p;
    };
    float* X   = alloc(MT * Dc);
    float* Qb  = alloc(MT * Dc);
    float* Kb  = alloc(MT * Dc);
    float* Vb  = alloc(MT * Dc);
    float* T1  = alloc(MT * FFc);
    float* KVb = alloc((size_t)Bc * Hc * DHc * DHc);
    float* Ksm = alloc((size_t)Bc * Hc * DHc);
    float* Zb  = alloc((size_t)Bc * Tc * Hc);
    float* CLSb= alloc((size_t)Bc * Dc);
    float* Y  = Qb;   // post-residual buffer (Q no longer live there)
    float* At = T1;   // attention output (FFN scratch not yet live)

    const int MG = (int)MT;
    dim3 blk(256);
    dim3 gEmb(Bc * SEQc / GBM, Dc / GBN);
    dim3 gD((MG + GBM - 1) / GBM, Dc / GBN);
    dim3 gF((MG + GBM - 1) / GBM, FFc / GBN);

    // Embedding: rows 1..T-1 of each batch, then the CLS row.
    gemm_wmma_kernel<<<gEmb, blk, 0, stream>>>(inputs, We, be, nullptr, pos, X,
                                               Bc * SEQc, Dc, Vc, EPI_EMBED);
    cls_embed_kernel<<<dim3((Bc * Dc + 255) / 256), blk, 0, stream>>>(We, be, pos, X);

    for (int l = 0; l < NLc; ++l) {
        const float* wq = Wq + (size_t)l * Dc * Dc;  const float* bql = bq + (size_t)l * Dc;
        const float* wk = Wk + (size_t)l * Dc * Dc;  const float* bkl = bk + (size_t)l * Dc;
        const float* wv = Wv + (size_t)l * Dc * Dc;  const float* bvl = bv + (size_t)l * Dc;
        const float* wo = Wo + (size_t)l * Dc * Dc;  const float* bol = bo + (size_t)l * Dc;
        const float* w1 = W1 + (size_t)l * Dc * FFc; const float* b1l = b1 + (size_t)l * FFc;
        const float* w2 = W2 + (size_t)l * FFc * Dc; const float* b2l = b2 + (size_t)l * Dc;

        gemm_wmma_kernel<<<gD, blk, 0, stream>>>(X, wq, bql, nullptr, nullptr, Qb, MG, Dc, Dc, EPI_BIAS);
        gemm_wmma_kernel<<<gD, blk, 0, stream>>>(X, wk, bkl, nullptr, nullptr, Kb, MG, Dc, Dc, EPI_BIAS);
        gemm_wmma_kernel<<<gD, blk, 0, stream>>>(X, wv, bvl, nullptr, nullptr, Vb, MG, Dc, Dc, EPI_BIAS);

        rope_elu_kernel<<<dim3((Bc * Tc * Hc * 32 + 255) / 256), blk, 0, stream>>>(Qb, Kb);
        ksum_kernel<<<dim3(Bc * Hc), blk, 0, stream>>>(Kb, Ksm);
        kv_wmma_kernel<<<dim3(Bc * Hc), dim3(128), 0, stream>>>(Kb, Vb, KVb);
        z_kernel<<<dim3((Bc * Tc * Hc + 255) / 256), blk, 0, stream>>>(Qb, Ksm, Zb);
        attn_wmma_kernel<<<dim3((Tc + 127) / 128, Bc * Hc), blk, 0, stream>>>(Qb, KVb, Zb, At);

        gemm_wmma_kernel<<<gD, blk, 0, stream>>>(At, wo, bol, X, nullptr, Y, MG, Dc, Dc, EPI_RES);
        ln_kernel<<<dim3(MG), blk, 0, stream>>>(Y, ln1g + (size_t)l * Dc, ln1b + (size_t)l * Dc, X, 1);

        gemm_wmma_kernel<<<gF, blk, 0, stream>>>(X, w1, b1l, nullptr, nullptr, T1, MG, FFc, Dc, EPI_GELU);
        gemm_wmma_kernel<<<gD, blk, 0, stream>>>(T1, w2, b2l, X, nullptr, Y, MG, Dc, FFc, EPI_RES);
        ln_kernel<<<dim3(MG), blk, 0, stream>>>(Y, ln2g + (size_t)l * Dc, ln2b + (size_t)l * Dc, X, 1);
    }

    // Final LN on CLS rows only, then the output head.
    ln_kernel<<<dim3(Bc), blk, 0, stream>>>(X, gN, bN, CLSb, Tc);
    head_kernel<<<dim3((Bc * Vc + 255) / 256), blk, 0, stream>>>(CLSb, Wout, bout, out);
}
